// GCN_14491219657218
// MI455X (gfx1250) — matrix operand order repaired and checked by reading, not compiled, not run on hardware
//
#include <hip/hip_runtime.h>

// ---- CDNA5 WMMA types ----
typedef __attribute__((ext_vector_type(16))) __bf16 v16bf;
typedef __attribute__((ext_vector_type(8)))  float  v8f;

#define D_IN  128
#define D_H   128
#define D_OUT 64

__device__ __forceinline__ void atom_add_f32(float* p, float v) {
  __hip_atomic_fetch_add(p, v, __ATOMIC_RELAXED, __HIP_MEMORY_SCOPE_AGENT);
}

// -------------------- degree / normalization --------------------

__global__ void init_deg(float* __restrict__ deg, int n) {
  int i = blockIdx.x * blockDim.x + threadIdx.x;
  if (i < n) deg[i] = 1.0f;  // self-loop contributes 1
}

__global__ void count_deg(const int* __restrict__ dst, float* __restrict__ deg, int E) {
  int e = blockIdx.x * blockDim.x + threadIdx.x;
  if (e < E) atom_add_f32(&deg[dst[e]], 1.0f);
}

__global__ void finalize_dinv(float* __restrict__ deg, int n) {
  int i = blockIdx.x * blockDim.x + threadIdx.x;
  if (i < n) deg[i] = rsqrtf(deg[i]);  // deg >= 1 always
}

// Wt[n*K + k] = (bf16) W[k*Ncols + n]   (W is K x Ncols row-major)
__global__ void transpose_to_bf16(const float* __restrict__ W, __bf16* __restrict__ Wt,
                                  int K, int Ncols) {
  int idx = blockIdx.x * blockDim.x + threadIdx.x;
  if (idx >= K * Ncols) return;
  int ncol = idx / K;
  int k    = idx - ncol * K;
  Wt[idx] = (__bf16)W[k * Ncols + ncol];
}

// -------------------- WMMA GEMM: C[M,N] = A[M,K] * Bt[N,K]^T --------------------
// Weights staged in LDS. One wave per 16-row M-tile; each wave sweeps all N tiles
// (NT accumulators) so every A fragment feeds NT back-to-back WMMAs.
// blockDim.x = 256 (8 waves = 8 M-tiles per block).
template <int K, int N>
__global__ void gemm_wmma_bf16(const float* __restrict__ A, const __bf16* __restrict__ Bt,
                               float* __restrict__ C, int M) {
  constexpr int NT = N / 16;   // N tiles per wave
  constexpr int KT = K / 32;   // K steps
  __shared__ __bf16 Bs[N * K];

  // cooperative load of Bt into LDS (N*K*2 bytes, 16B per thread per iter)
  {
    const uint4* gsrc = (const uint4*)Bt;
    uint4* ldst = (uint4*)Bs;
    const int nvec = (N * K * 2) / 16;
    for (int idx = threadIdx.x; idx < nvec; idx += blockDim.x) ldst[idx] = gsrc[idx];
  }
  __syncthreads();

  const int lane = threadIdx.x & 31;
  const int wave = threadIdx.x >> 5;
  const int lmod = lane & 15;
  const int lhi  = lane >> 4;  // 0: lanes 0-15, 1: lanes 16-31

  const int mbase = (blockIdx.x * (int)(blockDim.x >> 5) + wave) * 16;
  int mrow = mbase + lmod;
  if (mrow >= M) mrow = M - 1;  // clamp (keeps EXEC all-1 for WMMA)

  v8f acc[NT];
#pragma unroll
  for (int t = 0; t < NT; ++t) acc[t] = (v8f){};

  const float* arow = A + (size_t)mrow * K;

#pragma unroll
  for (int ks = 0; ks < KT; ++ks) {
    const int kb = ks * 32;
    // ---- A fragment: 16-bit A 16x32 layout ----
    // lanes 0-15 hold K = kb+0..7 and kb+16..23 ; lanes 16-31 hold kb+8..15, kb+24..31
    const float* ap = arow + kb + lhi * 8;
    float4 a0 = *(const float4*)(ap);
    float4 a1 = *(const float4*)(ap + 4);
    float4 a2 = *(const float4*)(ap + 16);
    float4 a3 = *(const float4*)(ap + 20);
    float av[16] = {a0.x, a0.y, a0.z, a0.w, a1.x, a1.y, a1.z, a1.w,
                    a2.x, a2.y, a2.z, a2.w, a3.x, a3.y, a3.z, a3.w};
    v16bf a;
#pragma unroll
    for (int j = 0; j < 16; ++j) a[j] = (__bf16)av[j];

    // ---- sweep all N tiles with this A fragment ----
#pragma unroll
    for (int t = 0; t < NT; ++t) {
      const __bf16* bp = Bs + (t * 16 + lmod) * K + kb + lhi * 8;
      typedef __attribute__((ext_vector_type(8))) __bf16 v8bf;
      v8bf b0 = *(const v8bf*)bp;
      v8bf b1 = *(const v8bf*)(bp + 16);
      v16bf b = __builtin_shufflevector(b0, b1, 0, 1, 2, 3, 4, 5, 6, 7, 8, 9, 10, 11, 12, 13, 14, 15);
      acc[t] = __builtin_amdgcn_wmma_f32_16x16x32_bf16(
          /*neg_a=*/false, a, /*neg_b=*/false, b,
          /*c_mod=*/(short)0, acc[t], /*reuse_a=*/false, /*reuse_b=*/false);
    }
  }

  // ---- C store: VGPR r -> row mbase + lhi*8 + r, col t*16 + lmod ----
  const int rbase = mbase + lhi * 8;
#pragma unroll
  for (int r = 0; r < 8; ++r) {
    const int row = rbase + r;
    if (row < M) {
      float* crow = C + (size_t)row * N + lmod;
#pragma unroll
      for (int t = 0; t < NT; ++t) crow[t * 16] = acc[t][r];
    }
  }
}

// -------------------- edge aggregation: out[dst] += h[src] * dinv[src]*dinv[dst] ----
// One wave per edge; D/32 features per lane. Atomic f32 adds (L2-resident).
template <int D>
__global__ void edge_agg(const int* __restrict__ src, const int* __restrict__ dst,
                         const float* __restrict__ dinv, const float* __restrict__ h,
                         float* __restrict__ out, int E) {
  constexpr int V = D / 32;
  int wave = blockIdx.x * (blockDim.x >> 5) + (threadIdx.x >> 5);
  if (wave >= E) return;
  int lane = threadIdx.x & 31;
  int s = src[wave];
  int d = dst[wave];
  float w = dinv[s] * dinv[d];
  const float* hp = h + (size_t)s * D + lane * V;
  float* op = out + (size_t)d * D + lane * V;
#pragma unroll
  for (int j = 0; j < V; ++j) atom_add_f32(op + j, hp[j] * w);
}

// -------------------- init aggregation buffer with self-loop + bias --------------------
// out[t] = h[t] * dinv[i]^2 + bias[f]   (the self-loop term of the segment sum)
template <int D>
__global__ void init_out(const float* __restrict__ h, const float* __restrict__ dinv,
                         const float* __restrict__ bias, float* __restrict__ out, int n) {
  long t = (long)blockIdx.x * blockDim.x + threadIdx.x;
  if (t >= (long)n * D) return;
  int i = (int)(t / D);
  int f = (int)(t - (long)i * D);
  float di = dinv[i];
  out[t] = h[t] * di * di + bias[f];
}

__global__ void relu_inplace(float* __restrict__ p, long n) {
  long i = (long)blockIdx.x * blockDim.x + threadIdx.x;
  long stride = (long)gridDim.x * blockDim.x;
  for (; i < n; i += stride) p[i] = fmaxf(p[i], 0.0f);
}

// -------------------- launcher --------------------
extern "C" void kernel_launch(void* const* d_in, const int* in_sizes, int n_in,
                              void* d_out, int out_size, void* d_ws, size_t ws_size,
                              hipStream_t stream) {
  const float* x  = (const float*)d_in[0];
  const int*   ei = (const int*)d_in[1];
  const float* W1 = (const float*)d_in[2];
  const float* b1 = (const float*)d_in[3];
  const float* W2 = (const float*)d_in[4];
  const float* b2 = (const float*)d_in[5];
  float* out = (float*)d_out;

  const int n = in_sizes[0] / D_IN;
  const int E = in_sizes[1] / 2;
  const int* src = ei;
  const int* dst = ei + E;

  // workspace carve-out
  char* ws = (char*)d_ws;
  size_t off = 0;
  auto carve = [&](size_t bytes) -> void* {
    void* p = ws + off;
    off += (bytes + 255) & ~(size_t)255;
    return p;
  };
  float*  dinv = (float*)carve((size_t)n * sizeof(float));
  __bf16* W1t  = (__bf16*)carve((size_t)D_H * D_IN * sizeof(__bf16));
  __bf16* W2t  = (__bf16*)carve((size_t)D_OUT * D_H * sizeof(__bf16));
  float*  h1   = (float*)carve((size_t)n * D_H * sizeof(float));
  float*  agg1 = (float*)carve((size_t)n * D_H * sizeof(float));
  float*  h2   = (float*)carve((size_t)n * D_OUT * sizeof(float));
  (void)ws_size; (void)n_in; (void)out_size;

  const int T = 256;
  const int mtiles = (n + 15) / 16;
  const int gemm_blocks = (mtiles + 7) / 8;  // 8 waves (M-tiles) per block

  // 1) symmetric normalization: dinv = rsqrt(1 + in-degree)
  init_deg<<<(n + T - 1) / T, T, 0, stream>>>(dinv, n);
  count_deg<<<(E + T - 1) / T, T, 0, stream>>>(dst, dinv, E);
  finalize_dinv<<<(n + T - 1) / T, T, 0, stream>>>(dinv, n);

  // 2) weights -> bf16, transposed (N rows of K contiguous)
  transpose_to_bf16<<<(D_H * D_IN + T - 1) / T, T, 0, stream>>>(W1, W1t, D_IN, D_H);
  transpose_to_bf16<<<(D_OUT * D_H + T - 1) / T, T, 0, stream>>>(W2, W2t, D_H, D_OUT);

  // 3) layer 1: h1 = x @ W1  (WMMA, weights in LDS)
  gemm_wmma_bf16<D_IN, D_H><<<gemm_blocks, T, 0, stream>>>(x, W1t, h1, n);

  // 4) aggregate layer 1: agg1 = selfloop + bias, then edge scatter, then ReLU
  init_out<D_H><<<(unsigned)(((long)n * D_H + T - 1) / T), T, 0, stream>>>(h1, dinv, b1, agg1, n);
  edge_agg<D_H><<<(E + 7) / 8, T, 0, stream>>>(src, dst, dinv, h1, agg1, E);
  relu_inplace<<<2048, T, 0, stream>>>(agg1, (long)n * D_H);

  // 5) layer 2: h2 = agg1 @ W2  (WMMA, weights in LDS)
  gemm_wmma_bf16<D_H, D_OUT><<<gemm_blocks, T, 0, stream>>>(agg1, W2t, h2, n);

  // 6) aggregate layer 2 directly into output
  init_out<D_OUT><<<(unsigned)(((long)n * D_OUT + T - 1) / T), T, 0, stream>>>(h2, dinv, b2, out, n);
  edge_agg<D_OUT><<<(E + 7) / 8, T, 0, stream>>>(src, dst, dinv, h2, out, E);
}
